// TAN_75239237091985
// MI455X (gfx1250) — compile-verified, matrix operand order checked by reading, not run on hardware
//
#include <hip/hip_runtime.h>
#include <math.h>

typedef __attribute__((ext_vector_type(16))) _Float16 v16h;
typedef __attribute__((ext_vector_type(8)))  float    v8f;
typedef __attribute__((ext_vector_type(4)))  unsigned u32x4;
typedef __attribute__((ext_vector_type(8)))  int      i32x8;
typedef __attribute__((ext_vector_type(4)))  int      i32x4;

#define BATCH     64
#define MANCH     7581
#define NCLS      80
#define MNC       (MANCH * NCLS)          // 606480 (divisible by 4)
#define KPRE      1000
#define KPAD      1024
#define TOPK_OUT  100
#define CAP       4096
#define NBINS     1024
#define GPB       1896                    // ceil(M/4) groups of 4 anchors per batch
#define TPW       4                       // tiles (WMMAs) per wave in decode
#define BPB       60                      // blocks per batch in decode: 60*8*4 = 1920 >= GPB
#define CONF_LOGIT (-2.9444389792f)       // logit(0.05)
#define NMS_T     0.6f
#define CLS_OFF   1216.0f                 // 2 * IMG per class (class-aware NMS offset)

// ---------------------------------------------------------------------------
// anchors: level 0: 76x76 stride 8, level 1: 38x38 stride 16, level 2: 19x19 stride 32
__device__ __forceinline__ void anchor_of(int m, float& ax, float& ay, float& st) {
    int n, base;
    if (m < 5776)      { n = 76; base = 0;    st = 8.0f;  }
    else if (m < 7220) { n = 38; base = 5776; st = 16.0f; }
    else               { n = 19; base = 7220; st = 32.0f; }
    int l = m - base;
    ax = (float)(l % n) + 0.5f;
    ay = (float)(l / n) + 0.5f;
}

__device__ __forceinline__ int bin_of(float x) {
    int bi = (int)((x + 8.0f) * 64.0f);
    return bi < 0 ? 0 : (bi > (NBINS - 1) ? (NBINS - 1) : bi);
}

// ---------------------------------------------------------------------------
// TDM: 1-D global -> LDS DMA descriptor (D#) per CDNA5 ISA 8.3/8.4.
// group0: count=1 | lds_addr | global_addr[56:0] | type=2 ("image")
// group1: data_size=4B, tensor_dim0=tile_dim0=n, tensor_dim1=1, stride=n
// groups 2/3 unused for a <=2-D tensor (zero-filled).
__device__ __forceinline__ void tdm_load_1d(const void* gptr, unsigned lds_off,
                                            unsigned n_elems) {
    const unsigned long long ga = (unsigned long long)(size_t)gptr;
    u32x4 g0;
    g0[0] = 1u;                                            // count=1 (user descriptor)
    g0[1] = lds_off;                                       // lds_addr (bytes)
    g0[2] = (unsigned)(ga & 0xFFFFFFFFu);                  // global_addr[31:0]
    g0[3] = ((unsigned)(ga >> 32) & 0x01FFFFFFu) | (2u << 30); // [56:32] | type=2
    i32x8 g1;
    g1[0] = (int)(2u << 16);                               // data_size=2 -> 4 bytes
    g1[1] = (int)((n_elems & 0xFFFFu) << 16);              // tensor_dim0[15:0] @ bit48
    g1[2] = (int)(((n_elems >> 16) & 0xFFFFu) | (1u << 16)); // dim0[31:16] | tensor_dim1=1
    g1[3] = (int)((n_elems & 0xFFFFu) << 16);              // tile_dim0 @ bit112
    g1[4] = 1;                                             // tile_dim1=1, tile_dim2=0
    g1[5] = (int)n_elems;                                  // tensor_dim0_stride[31:0]
    g1[6] = 0;                                             // stride[47:32] | dim1_stride lo
    g1[7] = 0;
    i32x4 z4 = {};
    i32x8 z8 = {};
    __builtin_amdgcn_tensor_load_to_lds(g0, g1, z4, z4, z8, 0);  // 6-arg form (clang-23)
}

// ---------------------------------------------------------------------------
// Kernel 0: zero histogram + per-batch candidate counters (ws is poisoned 0xAA)
__global__ void zero_ws_kernel(int* __restrict__ hist, int* __restrict__ ccount) {
    int i = blockIdx.x * blockDim.x + threadIdx.x;
    if (i < BATCH * NBINS) hist[i] = 0;
    if (i < BATCH) ccount[i] = 0;
}

// ---------------------------------------------------------------------------
// Kernel 1: DFL decode via WMMA, 4 tiles (= 16 anchors) per wave.
// Tile = 16 rows (4 anchors x 4 sides), 16 softmax bins per row.
// A (16x32 f16): lane L holds row (L&15), bins K[0..7] (half=L>>4 == 0) or
//                K[8..15] (half==1) in elements 0..7; elements 8..15 zero.
// B (32x16 f16): col 0 = ones (denominator), col 1 = arange (numerator).
// D (16x16 f32): cols 0/1 live on lanes 0,1,16,17 -> staged via wave-private LDS.
__global__ void decode_dfl_wmma(const float* __restrict__ reg,
                                float* __restrict__ boxes) {
    __shared__ float xg[8 * 32];   // 8 waves * (16 denom + 16 numer)
    const int w    = threadIdx.x >> 5;
    const int lane = threadIdx.x & 31;
    const int b    = blockIdx.x / BPB;
    const int wg0  = ((blockIdx.x - b * BPB) * 8 + w) * TPW;  // first group for this wave

    const int row  = lane & 15;
    const int half = lane >> 4;
    const int col  = lane & 15;

    // B fragment is loop-invariant
    v16h bv;
#pragma unroll
    for (int j = 0; j < 16; ++j) bv[j] = (_Float16)0.0f;
    if (col == 0) {
#pragma unroll
        for (int j = 0; j < 8; ++j) bv[j] = (_Float16)1.0f;
    } else if (col == 1) {
#pragma unroll
        for (int j = 0; j < 8; ++j) bv[j] = (_Float16)(float)(half * 8 + j);
    }

#pragma unroll
    for (int t = 0; t < TPW; ++t) {
        const int a0 = (wg0 + t) * 4;
        const int a  = a0 + (row >> 2);
        const int k  = row & 3;

        float vv[8];
        if (a < MANCH) {
            const float4* p = (const float4*)(reg + ((size_t)b * MANCH + a) * 64
                                                  + (size_t)k * 16 + (size_t)half * 8);
            float4 r0 = p[0], r1 = p[1];
            vv[0] = r0.x; vv[1] = r0.y; vv[2] = r0.z; vv[3] = r0.w;
            vv[4] = r1.x; vv[5] = r1.y; vv[6] = r1.z; vv[7] = r1.w;
        } else {
#pragma unroll
            for (int j = 0; j < 8; ++j) vv[j] = 0.0f;
        }

        // row max across lane pair (L, L^16), then exp(p - max) in f16
        float mx = vv[0];
#pragma unroll
        for (int j = 1; j < 8; ++j) mx = fmaxf(mx, vv[j]);
        mx = fmaxf(mx, __shfl_xor(mx, 16, 32));

        v16h av;
#pragma unroll
        for (int j = 0; j < 8; ++j) av[j] = (_Float16)__expf(vv[j] - mx);
#pragma unroll
        for (int j = 8; j < 16; ++j) av[j] = (_Float16)0.0f;

        v8f c = {};
        c = __builtin_amdgcn_wmma_f32_16x16x32_f16(false, av, false, bv,
                                                   (short)0, c, false, false);

        __syncthreads();   // previous iteration's readers are done with xg
        // D layout: VGPR v, lanes 0-15 -> (M=v, N=lane); lanes 16-31 -> (M=v+8, N=lane-16)
        // columns 0 (denom) / 1 (numer) live on lanes 0,1,16,17
        if (col < 2) {
            const int off = w * 32 + col * 16 + half * 8;
#pragma unroll
            for (int j = 0; j < 8; ++j) xg[off + j] = c[j];
        }
        __syncthreads();

        if (lane < 16) {
            const float den  = xg[w * 32 + lane];
            const float num  = xg[w * 32 + 16 + lane];
            const float dist = num / den;
            const int aa = a0 + (lane >> 2);
            const int kk = lane & 3;
            if (aa < MANCH) {
                float ax, ay, st;
                anchor_of(aa, ax, ay, st);
                const float base = (kk == 0 || kk == 2) ? ax : ay;
                const float val  = (kk < 2) ? (base - dist) : (base + dist);
                boxes[(((size_t)b * MANCH + aa) << 2) + kk] = val * st;
            }
        }
    }
}

// ---------------------------------------------------------------------------
// Kernel 2: per-batch logit histogram, b128 streaming (LDS-local, atomic flush)
__global__ void hist_kernel(const float* __restrict__ cls, int* __restrict__ hist) {
    __shared__ int lh[NBINS];
    const int b = blockIdx.y;
    for (int i = threadIdx.x; i < NBINS; i += blockDim.x) lh[i] = 0;
    __syncthreads();

    const float4* p = (const float4*)(cls + (size_t)b * MNC);
    const int n4 = MNC / 4;
    const int stride = gridDim.x * blockDim.x;
    for (int i = blockIdx.x * blockDim.x + threadIdx.x; i < n4; i += stride) {
        __builtin_prefetch(p + i + stride, 0, 1);   // global_prefetch_b8
        const float4 v = p[i];
        atomicAdd(&lh[bin_of(v.x)], 1);
        atomicAdd(&lh[bin_of(v.y)], 1);
        atomicAdd(&lh[bin_of(v.z)], 1);
        atomicAdd(&lh[bin_of(v.w)], 1);
    }
    __syncthreads();
    for (int i = threadIdx.x; i < NBINS; i += blockDim.x)
        if (lh[i]) atomicAdd(&hist[b * NBINS + i], lh[i]);
}

// ---------------------------------------------------------------------------
// Kernel 3: scan histogram from the top for the top-KPRE cutoff bin
__global__ void find_cutoff(const int* __restrict__ hist, int* __restrict__ cutoff) {
    const int b = threadIdx.x;
    if (b < BATCH) {
        int acc = 0, bin = 0;
        for (int i = NBINS - 1; i >= 0; --i) {
            acc += hist[b * NBINS + i];
            if (acc >= KPRE) { bin = i; break; }
        }
        cutoff[b] = bin;
    }
}

// ---------------------------------------------------------------------------
// Kernel 4: compact (logit, flat id) candidates with bin >= cutoff
__global__ void compact_candidates(const float* __restrict__ cls,
                                   const int* __restrict__ cutoff,
                                   int* __restrict__ ccount,
                                   float* __restrict__ cval,
                                   int* __restrict__ cid) {
    const int b = blockIdx.y;
    const int cut = cutoff[b];
    const float* p = cls + (size_t)b * MNC;
    const int stride = gridDim.x * blockDim.x;
    for (int i = blockIdx.x * blockDim.x + threadIdx.x; i < MNC; i += stride) {
        const float v = p[i];
        if (bin_of(v) >= cut) {
            const int pos = atomicAdd(&ccount[b], 1);
            if (pos < CAP) {
                cval[(size_t)b * CAP + pos] = v;
                cid[(size_t)b * CAP + pos]  = i;   // i = m*NC + class (memory order)
            }
        }
    }
}

// ---------------------------------------------------------------------------
// Kernel 5: per-batch in-LDS bitonic sort (descending; index tiebreak ascending).
// u64 key = monotone-transformed float bits (hi) | ~id (lo); invalid slots = 0.
__global__ void sort_candidates(const float* __restrict__ cval,
                                const int* __restrict__ cid,
                                const int* __restrict__ ccount,
                                float* __restrict__ sval,
                                int* __restrict__ sid) {
    __shared__ unsigned long long sk[CAP];
    const int b = blockIdx.x;
    int cnt = ccount[b];
    if (cnt > CAP) cnt = CAP;

    for (int i = threadIdx.x; i < CAP; i += blockDim.x) {
        unsigned long long key = 0ull;
        if (i < cnt) {
            unsigned u = __float_as_uint(cval[(size_t)b * CAP + i]);
            u = (u & 0x80000000u) ? ~u : (u | 0x80000000u);
            key = ((unsigned long long)u << 32)
                | (unsigned)(~(unsigned)cid[(size_t)b * CAP + i]);
        }
        sk[i] = key;
    }
    for (unsigned k = 2; k <= CAP; k <<= 1) {
        for (unsigned j = k >> 1; j > 0; j >>= 1) {
            __syncthreads();
            for (unsigned i = threadIdx.x; i < CAP; i += blockDim.x) {
                const unsigned ixj = i ^ j;
                if (ixj > i) {
                    unsigned long long A = sk[i], Bk = sk[ixj];
                    const bool blockAsc = ((i & k) == 0);
                    const bool sw = blockAsc ? (A < Bk) : (A > Bk);   // overall descending
                    if (sw) { sk[i] = Bk; sk[ixj] = A; }
                }
            }
        }
    }
    __syncthreads();
    for (int t = threadIdx.x; t < KPAD; t += blockDim.x) {
        const unsigned long long key = sk[t];
        const unsigned hi = (unsigned)(key >> 32), lo = (unsigned)key;
        const unsigned fb = (hi & 0x80000000u) ? (hi ^ 0x80000000u) : ~hi;
        sval[(size_t)b * KPAD + t] = __uint_as_float(fb);
        sid[(size_t)b * KPAD + t]  = (int)(~lo);
    }
}

// ---------------------------------------------------------------------------
// Kernel 6: per-batch class-offset greedy NMS over sorted top-1000 + top-100 output.
// Sorted candidate (val,id) arrays are DMA'd into LDS with the Tensor Data Mover.
__global__ void nms_finalize(const float* __restrict__ boxes,
                             const float* __restrict__ sval,
                             const int* __restrict__ sid,
                             const int* __restrict__ ccount,
                             float* __restrict__ out) {
    __shared__ float  sv_l[KPAD];
    __shared__ int    si_l[KPAD];
    __shared__ float4 bxs[KPRE];
    __shared__ float  vls[KPRE];
    __shared__ int    lbs[KPRE];
    __shared__ int    kp[KPRE];
    const int b = blockIdx.x;
    int cn = ccount[b];
    if (cn > CAP)  cn = CAP;
    if (cn > KPRE) cn = KPRE;

    // TDM bulk copy: 4KB vals + 4KB ids, global -> LDS (wave 0 issues; EXEC ignored)
    if (threadIdx.x == 0) {
        tdm_load_1d(sval + (size_t)b * KPAD, (unsigned)(size_t)&sv_l[0], KPAD);
        tdm_load_1d(sid  + (size_t)b * KPAD, (unsigned)(size_t)&si_l[0], KPAD);
    }
    __builtin_amdgcn_s_wait_tensorcnt(0);
    __syncthreads();

    for (int i = threadIdx.x; i < KPRE; i += blockDim.x) {
        if (i < cn) {
            const float v  = sv_l[i];
            const int   id = si_l[i];
            const int m   = id / NCLS;
            const int lab = id - m * NCLS;
            bxs[i] = *(const float4*)(boxes + (((size_t)b * MANCH + m) << 2));
            vls[i] = v; lbs[i] = lab;
            kp[i]  = (v > CONF_LOGIT) ? 1 : 0;
        } else {
            bxs[i] = make_float4(0.f, 0.f, 0.f, 0.f);
            vls[i] = 0.f; lbs[i] = 0; kp[i] = 0;
        }
    }
    __syncthreads();

    for (int i = 0; i < KPRE - 1; ++i) {
        if (kp[i]) {
            const float offi = (float)lbs[i] * CLS_OFF;
            const float4 A = bxs[i];
            const float ax1 = A.x + offi, ay1 = A.y + offi;
            const float ax2 = A.z + offi, ay2 = A.w + offi;
            const float areaA = fmaxf(ax2 - ax1, 0.f) * fmaxf(ay2 - ay1, 0.f);
            for (int j = i + 1 + threadIdx.x; j < KPRE; j += blockDim.x) {
                const float offj = (float)lbs[j] * CLS_OFF;
                const float4 Bx = bxs[j];
                const float bx1 = Bx.x + offj, by1 = Bx.y + offj;
                const float bx2 = Bx.z + offj, by2 = Bx.w + offj;
                const float areaB = fmaxf(bx2 - bx1, 0.f) * fmaxf(by2 - by1, 0.f);
                const float ix = fmaxf(fminf(ax2, bx2) - fmaxf(ax1, bx1), 0.f);
                const float iy = fmaxf(fminf(ay2, by2) - fmaxf(ay1, by1), 0.f);
                const float inter = ix * iy;
                const float iou = inter / (areaA + areaB - inter + 1e-9f);
                if (iou > NMS_T) kp[j] = 0;
            }
        }
        __syncthreads();
    }

    if (threadIdx.x == 0) {
        float* sc = out;
        float* lb = out + (size_t)BATCH * TOPK_OUT;
        float* bo = out + 2 * (size_t)BATCH * TOPK_OUT;
        int r = 0;
        for (int i = 0; i < KPRE && r < TOPK_OUT; ++i) if (kp[i]) {
            sc[b * TOPK_OUT + r] = 1.0f / (1.0f + __expf(-vls[i]));
            lb[b * TOPK_OUT + r] = (float)lbs[i];
            const float4 c = bxs[i];
            float* bp = bo + ((size_t)(b * TOPK_OUT + r)) * 4;
            bp[0] = c.x; bp[1] = c.y; bp[2] = c.z; bp[3] = c.w;
            ++r;
        }
        for (int i = 0; i < KPRE && r < TOPK_OUT; ++i) if (!kp[i]) {
            sc[b * TOPK_OUT + r] = 0.0f;
            lb[b * TOPK_OUT + r] = (float)lbs[i];   // reference leaves labels unmasked
            float* bp = bo + ((size_t)(b * TOPK_OUT + r)) * 4;
            bp[0] = 0.f; bp[1] = 0.f; bp[2] = 0.f; bp[3] = 0.f;
            ++r;
        }
    }
}

// ---------------------------------------------------------------------------
extern "C" void kernel_launch(void* const* d_in, const int* in_sizes, int n_in,
                              void* d_out, int out_size, void* d_ws, size_t ws_size,
                              hipStream_t stream) {
    (void)in_sizes; (void)n_in; (void)out_size; (void)ws_size;
    const float* reg = (const float*)d_in[0];
    const float* cls = (const float*)d_in[1];
    float* out = (float*)d_out;

    char* ws = (char*)d_ws;
    size_t o = 0;
    float* boxes  = (float*)(ws + o); o += (size_t)BATCH * MANCH * 4 * sizeof(float);
    int*   hist   = (int*)(ws + o);   o += (size_t)BATCH * NBINS * sizeof(int);
    int*   cutoff = (int*)(ws + o);   o += 256;
    int*   ccount = (int*)(ws + o);   o += 256;
    float* cval   = (float*)(ws + o); o += (size_t)BATCH * CAP * sizeof(float);
    int*   cid    = (int*)(ws + o);   o += (size_t)BATCH * CAP * sizeof(int);
    float* sval   = (float*)(ws + o); o += (size_t)BATCH * KPAD * sizeof(float);
    int*   sid    = (int*)(ws + o);   o += (size_t)BATCH * KPAD * sizeof(int);

    zero_ws_kernel<<<(BATCH * NBINS + 255) / 256, 256, 0, stream>>>(hist, ccount);
    decode_dfl_wmma<<<BATCH * BPB, 256, 0, stream>>>(reg, boxes);

    dim3 g2(64, BATCH);
    hist_kernel<<<g2, 256, 0, stream>>>(cls, hist);
    find_cutoff<<<1, BATCH, 0, stream>>>(hist, cutoff);
    compact_candidates<<<g2, 256, 0, stream>>>(cls, cutoff, ccount, cval, cid);
    sort_candidates<<<BATCH, 1024, 0, stream>>>(cval, cid, ccount, sval, sid);
    nms_finalize<<<BATCH, 256, 0, stream>>>(boxes, sval, sid, ccount, out);
}
